// GCN_32306744000869
// MI455X (gfx1250) — matrix operand order repaired and checked by reading, not compiled, not run on hardware
//
#include <hip/hip_runtime.h>
#include <hip/hip_bf16.h>

// ---------------------------------------------------------------------------
// GCN (3-layer, GCNConv-style) for MI455X / gfx1250.
//  - GEMMs via v_wmma_f32_16x16x32_bf16; one wave computes a 16-row x (NT*16)-col
//    strip so the A fragment is loaded once and fed to NT WMMAs (NT=4 for layer 1).
//  - Edge aggregation via hardware fp32 global atomics (targets fit in 192MB L2).
// Workspace usage: ~129.2 MB (see offsets below).
// ---------------------------------------------------------------------------

typedef __attribute__((ext_vector_type(16))) __bf16 v16bf;
typedef __attribute__((ext_vector_type(8)))  __bf16 v8bf;
typedef __attribute__((ext_vector_type(8)))  float  v8f;

__device__ __forceinline__ __bf16 f2bf(float f) {
  unsigned u = __builtin_bit_cast(unsigned, f);
  unsigned r = u + 0x7FFFu + ((u >> 16) & 1u);   // round-to-nearest-even
  return __builtin_bit_cast(__bf16, (unsigned short)(r >> 16));
}
__device__ __forceinline__ __bf16 bf0() {
  return __builtin_bit_cast(__bf16, (unsigned short)0);
}

// ----------------------------- WMMA GEMM -----------------------------------
// C[M,N] = A[M,Kp](bf16,row-major, Kp%32==0, zero-padded) @ B[Kp,N]
// B supplied transposed: BT[NT*16,Kp] (bf16, col-major of B, zero-padded).
// One wave computes a 16 x (NT*16) strip: A fragment loaded once per k-step,
// reused by NT WMMAs. Fragment layouts used here (A/B k order contiguous, matched):
//   A 16x32 bf16: lane L (row = L&15) holds K = kA..kA+15,
//                 kA = (L>>4)*16 -> two 16B loads.
//   B 32x16 bf16: lane L (col = L&15) holds K = kB..kB+15, kB = (L>>4)*16
//                 -> one 32B load.
//   C 16x16 f32:  lane L col = L&15, VGPR r -> row r (L<16) or 8+r (L>=16).
template <int NT>
__global__ __launch_bounds__(256) void gcn_wmma_gemm(
    const __bf16* __restrict__ A, const __bf16* __restrict__ BT,
    float* __restrict__ C, int M, int Kp, int N)
{
  const long long tiles_m = (long long)(M + 15) >> 4;
  const long long wave = (((long long)blockIdx.x * blockDim.x) + threadIdx.x) >> 5;
  if (wave >= tiles_m) return;            // wave-uniform: EXEC stays all-1
  const int lane = (int)(threadIdx.x & 31);
  const int tm = (int)wave;

  int rowA = tm * 16 + (lane & 15);
  if (rowA > M - 1) rowA = M - 1;         // clamp (M%16==0 in practice)
  const int kA = (lane >> 4) << 4;        // 0 or 16
  const int kB = (lane >> 4) << 4;        // 0 or 16

  const __bf16* arow = A  + (long long)rowA * Kp;
  const __bf16* bbase = BT + (long long)(lane & 15) * Kp + kB;

  v8f acc[NT] = {};
  for (int k = 0; k < Kp; k += 32) {
    union { v16bf v; v8bf h[2]; } ua;
    ua.h[0] = *(const v8bf*)(arow + k + kA);
    ua.h[1] = *(const v8bf*)(arow + k + kA + 8);
#pragma unroll
    for (int t = 0; t < NT; ++t) {
      const v16bf bv = *(const v16bf*)(bbase + (long long)t * 16 * Kp + k);
      acc[t] = __builtin_amdgcn_wmma_f32_16x16x32_bf16(
          /*neg_a=*/false, ua.v, /*neg_b=*/false, bv,
          /*c_mod=*/(short)0, acc[t], /*reuse_a=*/false, /*reuse_b=*/false);
    }
  }

  const int rbase = tm * 16 + ((lane >> 4) << 3);
#pragma unroll
  for (int t = 0; t < NT; ++t) {
    const int col = t * 16 + (lane & 15);
    if (col < N) {
#pragma unroll
      for (int r = 0; r < 8; ++r) {
        const int row = rbase + r;
        if (row < M) C[(long long)row * N + col] = acc[t][r];
      }
    }
  }
}

// ----------------------------- pack kernels --------------------------------
__global__ __launch_bounds__(256) void gcn_pack_rows(
    const float* __restrict__ in, __bf16* __restrict__ out,
    long long M, int K, int Kp)
{
  const long long total = M * Kp;
  const long long stride = (long long)gridDim.x * blockDim.x;
  for (long long i = (long long)blockIdx.x * blockDim.x + threadIdx.x;
       i < total; i += stride) {
    const long long n = i / Kp;
    const int k = (int)(i - n * Kp);
    out[i] = (k < K) ? f2bf(in[n * K + k]) : bf0();
  }
}

// W[K,N] (row-major) -> BT[Npad,Kp] column-major, zero-padded
__global__ __launch_bounds__(256) void gcn_packT(
    const float* __restrict__ W, __bf16* __restrict__ out,
    int K, int N, int Kp, int Npad)
{
  const int total = Npad * Kp;
  const int stride = (int)(gridDim.x * blockDim.x);
  for (int i = (int)(blockIdx.x * blockDim.x + threadIdx.x);
       i < total; i += stride) {
    const int n = i / Kp;
    const int k = i - n * Kp;
    out[i] = (n < N && k < K) ? f2bf(W[(long long)k * N + n]) : bf0();
  }
}

// ----------------------------- degree / norm -------------------------------
__global__ __launch_bounds__(256) void gcn_deg_init(float* deg, long long M) {
  const long long stride = (long long)gridDim.x * blockDim.x;
  for (long long i = (long long)blockIdx.x * blockDim.x + threadIdx.x;
       i < M; i += stride)
    deg[i] = 1.0f;                                // self-loop
}
__global__ __launch_bounds__(256) void gcn_deg_edges(
    const long long* __restrict__ ei, float* deg, long long E)
{
  const long long stride = (long long)gridDim.x * blockDim.x;
  for (long long e = (long long)blockIdx.x * blockDim.x + threadIdx.x;
       e < E; e += stride)
    unsafeAtomicAdd(&deg[(int)ei[E + e]], 1.0f);  // dst row
}
__global__ __launch_bounds__(256) void gcn_deg_rsqrt(float* deg, long long M) {
  const long long stride = (long long)gridDim.x * blockDim.x;
  for (long long i = (long long)blockIdx.x * blockDim.x + threadIdx.x;
       i < M; i += stride)
    deg[i] = rsqrtf(deg[i]);                      // deg >= 1 always
}

// ----------------------------- aggregation ---------------------------------
// a[n,:] = t[n,:]*dinv[n]^2  (self-loop contribution as initializer)
__global__ __launch_bounds__(256) void gcn_self_init(
    const float* __restrict__ t, const float* __restrict__ dinv,
    float* __restrict__ a, long long M, int F)
{
  const long long total = M * F;
  const long long stride = (long long)gridDim.x * blockDim.x;
  for (long long i = (long long)blockIdx.x * blockDim.x + threadIdx.x;
       i < total; i += stride) {
    const float di = dinv[i / F];
    a[i] = t[i] * di * di;
  }
}

// a[dst,:] += t[src,:] * dinv[src]*dinv[dst], CH features per thread
template <int F, int CH>
__global__ __launch_bounds__(256) void gcn_edge_agg(
    const long long* __restrict__ ei, const float* __restrict__ dinv,
    const float* __restrict__ t, float* __restrict__ a, long long E)
{
  constexpr long long NCH = F / CH;
  const long long total = E * NCH;
  const long long stride = (long long)gridDim.x * blockDim.x;
  for (long long i = (long long)blockIdx.x * blockDim.x + threadIdx.x;
       i < total; i += stride) {
    const long long e = i / NCH;
    const int c = (int)(i - e * NCH) * CH;
    const int s = (int)ei[e];
    const int d = (int)ei[E + e];
    const float nrm = dinv[s] * dinv[d];
    const float* ts = t + (long long)s * F + c;
    float* ad = a + (long long)d * F + c;
    if constexpr (CH % 4 == 0) {
#pragma unroll
      for (int f = 0; f < CH; f += 4) {
        const float4 v = *(const float4*)(ts + f);
        unsafeAtomicAdd(ad + f + 0, v.x * nrm);
        unsafeAtomicAdd(ad + f + 1, v.y * nrm);
        unsafeAtomicAdd(ad + f + 2, v.z * nrm);
        unsafeAtomicAdd(ad + f + 3, v.w * nrm);
      }
    } else {
#pragma unroll
      for (int f = 0; f < CH; ++f)
        unsafeAtomicAdd(ad + f, ts[f] * nrm);
    }
  }
}

// ----------------------------- epilogues -----------------------------------
__global__ __launch_bounds__(256) void gcn_bias_relu(
    float* __restrict__ a, const float* __restrict__ b, long long total, int F)
{
  const long long stride = (long long)gridDim.x * blockDim.x;
  for (long long i = (long long)blockIdx.x * blockDim.x + threadIdx.x;
       i < total; i += stride) {
    const int f = (int)(i % F);
    a[i] = fmaxf(a[i] + b[f], 0.0f);
  }
}

__global__ __launch_bounds__(256) void gcn_logsm2(
    const float* __restrict__ a, const float* __restrict__ b,
    float* __restrict__ out, long long M)
{
  const long long stride = (long long)gridDim.x * blockDim.x;
  for (long long n = (long long)blockIdx.x * blockDim.x + threadIdx.x;
       n < M; n += stride) {
    const float z0 = a[2 * n + 0] + b[0];
    const float z1 = a[2 * n + 1] + b[1];
    const float m = fmaxf(z0, z1);
    const float l = m + logf(expf(z0 - m) + expf(z1 - m));
    out[2 * n + 0] = z0 - l;
    out[2 * n + 1] = z1 - l;
  }
}

// ----------------------------- host side -----------------------------------
static inline unsigned gblk(long long work) {
  long long b = (work + 255) / 256;
  if (b > 0x7FFFFFFLL) b = 0x7FFFFFF;   // grid-stride loops absorb the cap
  return (unsigned)b;
}

extern "C" void kernel_launch(void* const* d_in, const int* in_sizes, int n_in,
                              void* d_out, int out_size, void* d_ws, size_t ws_size,
                              hipStream_t stream) {
  const float*      x  = (const float*)d_in[0];
  const long long*  ei = (const long long*)d_in[1];   // [2,E] int64
  const float* W1 = (const float*)d_in[2];
  const float* b1 = (const float*)d_in[3];
  const float* W2 = (const float*)d_in[4];
  const float* b2 = (const float*)d_in[5];
  const float* W3 = (const float*)d_in[6];
  const float* b3 = (const float*)d_in[7];
  float* out = (float*)d_out;

  const long long M = (long long)in_sizes[0] / 165;   // 200000
  const long long E = (long long)in_sizes[1] / 2;     // 3200000

  // ---- workspace layout (bytes), all offsets 256-aligned; total ~129.2 MB --
  char* ws = (char*)d_ws;
  float*  dinv = (float*)(ws + 0);                          // 0.8 MB
  const size_t OW = 1u << 20;
  __bf16* wT1 = (__bf16*)(ws + OW);                         // 64x192 bf16
  __bf16* wT2 = (__bf16*)(ws + OW + 32768);                 // 16x64  bf16
  __bf16* wT3 = (__bf16*)(ws + OW + 40960);                 // 16x32  bf16
  char* R1 = ws + OW + 65536;                               // 76.8 MB region
  char* R2 = R1 + 76800000;                                 // 51.2 MB region
  __bf16* x_bf  = (__bf16*)R1;              // phase 1-2 (M x 192)
  float*  t1    = (float*)R2;               // M x 64
  float*  h1    = (float*)R1;               // M x 64 (overwrites x_bf)
  __bf16* h1_bf = (__bf16*)R2;              // M x 64 (overwrites t1)
  float*  t2    = (float*)(R2 + 25600000);  // M x 16
  float*  a2    = (float*)(R1 + 51200000);  // M x 16
  __bf16* h2_bf = (__bf16*)R2;              // M x 32 (overwrites h1_bf)
  float*  t3    = (float*)(R2 + 12800000);  // M x 2
  float*  a3    = (float*)R1;               // M x 2  (overwrites h1)

  const long long tiles_m = (M + 15) / 16;

  // ---- weight packs (tiny) ----
  gcn_packT<<<gblk(64 * 192), 256, 0, stream>>>(W1, wT1, 165, 64, 192, 64);
  gcn_packT<<<gblk(16 * 64),  256, 0, stream>>>(W2, wT2, 64, 16, 64, 16);
  gcn_packT<<<gblk(16 * 32),  256, 0, stream>>>(W3, wT3, 16, 2, 32, 16);

  // ---- degrees -> dinv ----
  gcn_deg_init <<<gblk(M), 256, 0, stream>>>(dinv, M);
  gcn_deg_edges<<<gblk(E), 256, 0, stream>>>(ei, dinv, E);
  gcn_deg_rsqrt<<<gblk(M), 256, 0, stream>>>(dinv, M);

  // ---- layer 1: 165 -> 64 (one wave covers all 4 column tiles) ----
  gcn_pack_rows<<<gblk(M * 192), 256, 0, stream>>>(x, x_bf, M, 165, 192);
  gcn_wmma_gemm<4><<<gblk(tiles_m * 32), 256, 0, stream>>>(x_bf, wT1, t1,
                                                           (int)M, 192, 64);
  gcn_self_init<<<gblk(M * 64), 256, 0, stream>>>(t1, dinv, h1, M, 64);
  gcn_edge_agg<64, 16><<<gblk(E * 4), 256, 0, stream>>>(ei, dinv, t1, h1, E);
  gcn_bias_relu<<<gblk(M * 64), 256, 0, stream>>>(h1, b1, M * 64, 64);

  // ---- layer 2: 64 -> 16 ----
  gcn_pack_rows<<<gblk(M * 64), 256, 0, stream>>>(h1, h1_bf, M, 64, 64);
  gcn_wmma_gemm<1><<<gblk(tiles_m * 32), 256, 0, stream>>>(h1_bf, wT2, t2,
                                                           (int)M, 64, 16);
  gcn_self_init<<<gblk(M * 16), 256, 0, stream>>>(t2, dinv, a2, M, 16);
  gcn_edge_agg<16, 16><<<gblk(E), 256, 0, stream>>>(ei, dinv, t2, a2, E);
  gcn_bias_relu<<<gblk(M * 16), 256, 0, stream>>>(a2, b2, M * 16, 16);

  // ---- layer 3: 16 -> 2 ----
  gcn_pack_rows<<<gblk(M * 32), 256, 0, stream>>>(a2, h2_bf, M, 16, 32);
  gcn_wmma_gemm<1><<<gblk(tiles_m * 32), 256, 0, stream>>>(h2_bf, wT3, t3,
                                                           (int)M, 32, 2);
  gcn_self_init<<<gblk(M * 2), 256, 0, stream>>>(t3, dinv, a3, M, 2);
  gcn_edge_agg<2, 2><<<gblk(E), 256, 0, stream>>>(ei, dinv, t3, a3, E);

  // ---- log-softmax (bias fused) ----
  gcn_logsm2<<<gblk(M), 256, 0, stream>>>(a3, b3, out, M);
}
